// SVDHead_48833778155887
// MI455X (gfx1250) — matrix operand order repaired and checked by reading, not compiled, not run on hardware
//
#include <hip/hip_runtime.h>
#include <hip/hip_bf16.h>
#include <math.h>

typedef __attribute__((ext_vector_type(16))) _Float16 v16h;
typedef __attribute__((ext_vector_type(8)))  _Float16 v8h;
typedef __attribute__((ext_vector_type(8)))  float    v8f;

// ---------------------------------------------------------------------------
// Transpose+convert: in [B][D][P] f32  ->  out [B][P][D] f16  (K-contiguous rows)
// ---------------------------------------------------------------------------
__global__ void k_transpose_f16(const float* __restrict__ in, _Float16* __restrict__ out,
                                int D, int P) {
    __shared__ _Float16 tile[32][33];
    const int b  = blockIdx.z;
    const int p0 = blockIdx.x * 32;
    const int d0 = blockIdx.y * 32;
    const float*  src = in  + (size_t)b * D * P;
    _Float16*     dst = out + (size_t)b * P * D;
    #pragma unroll
    for (int i = 0; i < 4; ++i) {
        int d = d0 + threadIdx.y + i * 8;
        int p = p0 + threadIdx.x;
        tile[threadIdx.y + i * 8][threadIdx.x] = (_Float16)src[(size_t)d * P + p];
    }
    __syncthreads();
    #pragma unroll
    for (int i = 0; i < 4; ++i) {
        int p = p0 + threadIdx.y + i * 8;
        int d = d0 + threadIdx.x;
        dst[(size_t)p * D + d] = tile[threadIdx.x][threadIdx.y + i * 8];
    }
}

__global__ void k_zero_f32(float* __restrict__ p, int n) {
    int i = blockIdx.x * blockDim.x + threadIdx.x;
    if (i < n) p[i] = 0.0f;
}

// ---------------------------------------------------------------------------
// WMMA GEMM: S[b][n][m] = scale * sum_k A16[b][n][k] * B16[b][m][k]
// Each wave: one 32x64 strip (2 A frags x 4 B frags -> 8 WMMAs per K-step of 32;
// B frags reused across both row tiles: 12 b128 loads per 8 WMMAs).
// Fragment layouts per CDNA5 ISA 7.12.2 (16-bit A 16x32, B 32x16, f32 C/D 16x16).
// ---------------------------------------------------------------------------
__global__ void __launch_bounds__(256)
k_gemm_wmma(const _Float16* __restrict__ A, const _Float16* __restrict__ Bm,
            float* __restrict__ S, int N, int M, int K, float scale) {
    const int b    = blockIdx.z;
    const int lane = threadIdx.x & 31;
    const int wave = threadIdx.x >> 5;
    const int h    = lane >> 4;     // K-half selector
    const int l16  = lane & 15;     // row (A) / col (B,D) index
    const int n0   = blockIdx.y * 256 + wave * 32;
    const int m0   = blockIdx.x * 64;

    const _Float16* Ab = A  + (size_t)b * N * K;
    const _Float16* Bb = Bm + (size_t)b * M * K;
    float*          Sb = S  + (size_t)b * N * M;

    const _Float16* arow0 = Ab + (size_t)(n0 + l16) * K;
    const _Float16* arow1 = Ab + (size_t)(n0 + 16 + l16) * K;
    const _Float16* brow0 = Bb + (size_t)(m0 + 0  + l16) * K;
    const _Float16* brow1 = Bb + (size_t)(m0 + 16 + l16) * K;
    const _Float16* brow2 = Bb + (size_t)(m0 + 32 + l16) * K;
    const _Float16* brow3 = Bb + (size_t)(m0 + 48 + l16) * K;

    v8f acc[2][4];
    #pragma unroll
    for (int ti = 0; ti < 2; ++ti)
        #pragma unroll
        for (int j = 0; j < 4; ++j)
            #pragma unroll
            for (int e = 0; e < 8; ++e) acc[ti][j][e] = 0.0f;

    for (int kk = 0; kk < K; kk += 32) {
        // A frags: lanes 0-15 K=0..15, lanes 16-31 K=16..31 (contiguous per half)
        v8h a0lo = *(const v8h*)(arow0 + kk + h * 16);
        v8h a0hi = *(const v8h*)(arow0 + kk + h * 16 + 8);
        v8h a1lo = *(const v8h*)(arow1 + kk + h * 16);
        v8h a1hi = *(const v8h*)(arow1 + kk + h * 16 + 8);
        v16h a0 = __builtin_shufflevector(a0lo, a0hi,
                   0, 1, 2, 3, 4, 5, 6, 7, 8, 9, 10, 11, 12, 13, 14, 15);
        v16h a1 = __builtin_shufflevector(a1lo, a1hi,
                   0, 1, 2, 3, 4, 5, 6, 7, 8, 9, 10, 11, 12, 13, 14, 15);
        // B frags: lanes 0-15 K=0..15, lanes 16-31 K=16..31 (contiguous per half)
        v16h b0 = *(const v16h*)(brow0 + kk + h * 16);
        v16h b1 = *(const v16h*)(brow1 + kk + h * 16);
        v16h b2 = *(const v16h*)(brow2 + kk + h * 16);
        v16h b3 = *(const v16h*)(brow3 + kk + h * 16);
        acc[0][0] = __builtin_amdgcn_wmma_f32_16x16x32_f16(false, a0, false, b0, (short)0, acc[0][0], false, false);
        acc[0][1] = __builtin_amdgcn_wmma_f32_16x16x32_f16(false, a0, false, b1, (short)0, acc[0][1], false, false);
        acc[0][2] = __builtin_amdgcn_wmma_f32_16x16x32_f16(false, a0, false, b2, (short)0, acc[0][2], false, false);
        acc[0][3] = __builtin_amdgcn_wmma_f32_16x16x32_f16(false, a0, false, b3, (short)0, acc[0][3], false, false);
        acc[1][0] = __builtin_amdgcn_wmma_f32_16x16x32_f16(false, a1, false, b0, (short)0, acc[1][0], false, false);
        acc[1][1] = __builtin_amdgcn_wmma_f32_16x16x32_f16(false, a1, false, b1, (short)0, acc[1][1], false, false);
        acc[1][2] = __builtin_amdgcn_wmma_f32_16x16x32_f16(false, a1, false, b2, (short)0, acc[1][2], false, false);
        acc[1][3] = __builtin_amdgcn_wmma_f32_16x16x32_f16(false, a1, false, b3, (short)0, acc[1][3], false, false);
    }
    // D layout: element r of lane -> row tileRow + r + 8*h, col m0 + 16*j + l16
    #pragma unroll
    for (int ti = 0; ti < 2; ++ti)
        #pragma unroll
        for (int j = 0; j < 4; ++j)
            #pragma unroll
            for (int r = 0; r < 8; ++r)
                Sb[(size_t)(n0 + ti * 16 + r + 8 * h) * M + (m0 + j * 16 + l16)] =
                    acc[ti][j][r] * scale;
}

// ---------------------------------------------------------------------------
// Block reductions (blockDim.x == 256)
// ---------------------------------------------------------------------------
__device__ __forceinline__ float block_sum(float v, float* red) {
    int t = threadIdx.x;
    red[t] = v; __syncthreads();
    for (int s = 128; s > 0; s >>= 1) { if (t < s) red[t] += red[t + s]; __syncthreads(); }
    float r = red[0]; __syncthreads();
    return r;
}
__device__ __forceinline__ float block_max(float v, float* red) {
    int t = threadIdx.x;
    red[t] = v; __syncthreads();
    for (int s = 128; s > 0; s >>= 1) { if (t < s) red[t] = fmaxf(red[t], red[t + s]); __syncthreads(); }
    float r = red[0]; __syncthreads();
    return r;
}

// ---------------------------------------------------------------------------
// Sinkhorn row normalization: rows 0..N-1 over M real cols + slack col c[b][n]
// one block per (b, n)
// ---------------------------------------------------------------------------
__global__ void __launch_bounds__(256)
k_rownorm(float* __restrict__ S, float* __restrict__ cslack, int N, int M) {
    __shared__ float red[256];
    const int b = blockIdx.y, n = blockIdx.x, t = threadIdx.x;
    float* row = S + ((size_t)b * N + n) * M;
    float* cs  = cslack + (size_t)b * N + n;
    float mx = -1e30f;
    for (int m = t; m < M; m += 256) mx = fmaxf(mx, row[m]);
    if (t == 0) mx = fmaxf(mx, *cs);
    mx = block_max(mx, red);
    float sum = 0.0f;
    for (int m = t; m < M; m += 256) sum += __expf(row[m] - mx);
    if (t == 0) sum += __expf(*cs - mx);
    sum = block_sum(sum, red);
    float L = mx + __logf(sum);
    for (int m = t; m < M; m += 256) row[m] -= L;
    if (t == 0) *cs -= L;
}

// ---------------------------------------------------------------------------
// Sinkhorn col normalization: cols 0..M-1 over N real rows + slack row r[b][m]
// one thread per column, online logsumexp (coalesced across the block)
// ---------------------------------------------------------------------------
__global__ void __launch_bounds__(256)
k_colnorm(float* __restrict__ S, float* __restrict__ rslack, int N, int M) {
    const int b = blockIdx.y;
    const int m = blockIdx.x * 256 + threadIdx.x;
    float* Sb = S + (size_t)b * N * M;
    float* rs = rslack + (size_t)b * M + m;
    float mx = *rs;          // slack entry seeds the running logsumexp
    float sum = 1.0f;        // exp(rs - rs)
    for (int n = 0; n < N; ++n) {
        float x = Sb[(size_t)n * M + m];
        if (x > mx) { sum = sum * __expf(mx - x) + 1.0f; mx = x; }
        else        { sum += __expf(x - mx); }
    }
    float L = mx + __logf(sum);
    for (int n = 0; n < N; ++n) Sb[(size_t)n * M + m] -= L;
    *rs -= L;
}

// ---------------------------------------------------------------------------
// perm = exp(S); perm_norm -> d_out; weights = rowmax(perm);
// weighted_tgt[b][d][n] = sum_m tgt[b][d][m] * perm_norm[n][m]
// one block per (b, n)
// ---------------------------------------------------------------------------
__global__ void __launch_bounds__(256)
k_finalize_rows(const float* __restrict__ S, const float* __restrict__ tgt,
                float* __restrict__ perm_out, float* __restrict__ wtgt,
                float* __restrict__ wmax, int N, int M) {
    __shared__ float red[256];
    const int b = blockIdx.y, n = blockIdx.x, t = threadIdx.x;
    const float* row = S + ((size_t)b * N + n) * M;
    const float* tg  = tgt + (size_t)b * 3 * M;
    float mx = -1e30f, sum = 0.0f;
    for (int m = t; m < M; m += 256) {
        float e = __expf(row[m]);
        sum += e; mx = fmaxf(mx, e);
    }
    float rowsum = block_sum(sum, red);
    float rowmax = block_max(mx, red);
    float inv = 1.0f / (rowsum + 1e-8f);
    float a0 = 0.0f, a1 = 0.0f, a2 = 0.0f;
    float* po = perm_out + ((size_t)b * N + n) * M;
    for (int m = t; m < M; m += 256) {
        float pn = __expf(row[m]) * inv;
        po[m] = pn;
        a0 += pn * tg[m];
        a1 += pn * tg[(size_t)M + m];
        a2 += pn * tg[(size_t)2 * M + m];
    }
    a0 = block_sum(a0, red);
    a1 = block_sum(a1, red);
    a2 = block_sum(a2, red);
    if (t == 0) {
        wtgt[((size_t)b * 3 + 0) * N + n] = a0;
        wtgt[((size_t)b * 3 + 1) * N + n] = a1;
        wtgt[((size_t)b * 3 + 2) * N + n] = a2;
        wmax[(size_t)b * N + n] = rowmax;
    }
}

// ---------------------------------------------------------------------------
// Per-batch: centroids, H (3x3), one-sided Jacobi 3x3 SVD, R, t
// ---------------------------------------------------------------------------
__device__ __forceinline__ float det3(const float m[3][3]) {
    return m[0][0] * (m[1][1] * m[2][2] - m[1][2] * m[2][1])
         - m[0][1] * (m[1][0] * m[2][2] - m[1][2] * m[2][0])
         + m[0][2] * (m[1][0] * m[2][1] - m[1][1] * m[2][0]);
}

__global__ void __launch_bounds__(256)
k_svd_head(const float* __restrict__ src, const float* __restrict__ wtgt,
           const float* __restrict__ wmax, float* __restrict__ outR,
           float* __restrict__ outT, int N) {
    __shared__ float red[256];
    __shared__ float sh[16];
    const int b = blockIdx.x, t = threadIdx.x;
    const float* sp = src  + (size_t)b * 3 * N;
    const float* wp = wtgt + (size_t)b * 3 * N;
    const float* w  = wmax + (size_t)b * N;

    float s = 0.0f;
    for (int n = t; n < N; n += 256) s += w[n];
    float Wsum = block_sum(s, red);
    float winv = 1.0f / (Wsum + 1e-8f);

    for (int d = 0; d < 3; ++d) {
        float a = 0.0f, c = 0.0f;
        for (int n = t; n < N; n += 256) {
            float wn = w[n] * winv;
            a += sp[(size_t)d * N + n] * wn;
            c += wp[(size_t)d * N + n] * wn;
        }
        float as = block_sum(a, red);
        float cs = block_sum(c, red);
        if (t == 0) { sh[d] = as; sh[3 + d] = cs; }
    }
    __syncthreads();
    const float sc[3] = { sh[0], sh[1], sh[2] };
    const float tc[3] = { sh[3], sh[4], sh[5] };

    for (int ij = 0; ij < 9; ++ij) {
        int i = ij / 3, j = ij % 3;
        float sci = sc[i], tcj = tc[j];
        float a = 0.0f;
        for (int n = t; n < N; n += 256) {
            float wn = w[n] * winv;
            a += (sp[(size_t)i * N + n] - sci) * ((wp[(size_t)j * N + n] - tcj) * wn);
        }
        float hs = block_sum(a, red);
        if (t == 0) sh[6 + ij] = hs;
    }
    __syncthreads();

    if (t == 0) {
        float Amat[3][3], V[3][3] = {{1, 0, 0}, {0, 1, 0}, {0, 0, 1}};
        for (int i = 0; i < 3; ++i)
            for (int j = 0; j < 3; ++j) Amat[i][j] = sh[6 + i * 3 + j];
        // one-sided Jacobi: H * (J1 J2 ...) = U*Sigma, V = prod(Jk)
        for (int sw = 0; sw < 15; ++sw) {
            for (int pi = 0; pi < 3; ++pi) {
                const int p = (pi == 2) ? 1 : 0;
                const int q = (pi == 0) ? 1 : 2;
                float app = 0, aqq = 0, apq = 0;
                for (int r = 0; r < 3; ++r) {
                    app += Amat[r][p] * Amat[r][p];
                    aqq += Amat[r][q] * Amat[r][q];
                    apq += Amat[r][p] * Amat[r][q];
                }
                if (fabsf(apq) > 1e-20f) {
                    float tau = (aqq - app) / (2.0f * apq);
                    float tt  = ((tau >= 0.0f) ? 1.0f : -1.0f) /
                                (fabsf(tau) + sqrtf(1.0f + tau * tau));
                    float cc  = 1.0f / sqrtf(1.0f + tt * tt);
                    float ss  = tt * cc;
                    for (int r = 0; r < 3; ++r) {
                        float ap = Amat[r][p], aq = Amat[r][q];
                        Amat[r][p] = cc * ap - ss * aq;
                        Amat[r][q] = ss * ap + cc * aq;
                        float vp = V[r][p], vq = V[r][q];
                        V[r][p] = cc * vp - ss * vq;
                        V[r][q] = ss * vp + cc * vq;
                    }
                }
            }
        }
        float sig[3];
        for (int c = 0; c < 3; ++c)
            sig[c] = sqrtf(Amat[0][c] * Amat[0][c] + Amat[1][c] * Amat[1][c] +
                           Amat[2][c] * Amat[2][c]);
        int idx[3] = {0, 1, 2};
        for (int i = 0; i < 2; ++i)
            for (int j = i + 1; j < 3; ++j)
                if (sig[idx[j]] > sig[idx[i]]) { int tmp = idx[i]; idx[i] = idx[j]; idx[j] = tmp; }
        float U[3][3], Vs[3][3];
        for (int c = 0; c < 3; ++c) {
            int k = idx[c];
            float inv = (sig[k] > 1e-20f) ? 1.0f / sig[k] : 0.0f;
            for (int r = 0; r < 3; ++r) { U[r][c] = Amat[r][k] * inv; Vs[r][c] = V[r][k]; }
        }
        if (sig[idx[2]] <= 1e-20f) {  // complete U with a right-handed third column
            float cx = U[1][0] * U[2][1] - U[2][0] * U[1][1];
            float cy = U[2][0] * U[0][1] - U[0][0] * U[2][1];
            float cz = U[0][0] * U[1][1] - U[1][0] * U[0][1];
            float nn = sqrtf(cx * cx + cy * cy + cz * cz);
            if (nn > 1e-20f) { U[0][2] = cx / nn; U[1][2] = cy / nn; U[2][2] = cz / nn; }
            else             { U[0][2] = 0.0f;    U[1][2] = 0.0f;    U[2][2] = 1.0f;    }
        }
        float det = det3(Vs) * det3(U);   // det(V U^T)
        float R[3][3];
        for (int i = 0; i < 3; ++i)
            for (int j = 0; j < 3; ++j)
                R[i][j] = Vs[i][0] * U[j][0] + Vs[i][1] * U[j][1] + det * Vs[i][2] * U[j][2];
        for (int i = 0; i < 3; ++i) {
            for (int j = 0; j < 3; ++j) outR[(size_t)b * 9 + i * 3 + j] = R[i][j];
            outT[(size_t)b * 3 + i] =
                -(R[i][0] * sc[0] + R[i][1] * sc[1] + R[i][2] * sc[2]) + tc[i];
        }
    }
}

// ---------------------------------------------------------------------------
extern "C" void kernel_launch(void* const* d_in, const int* in_sizes, int n_in,
                              void* d_out, int out_size, void* d_ws, size_t ws_size,
                              hipStream_t stream) {
    (void)n_in; (void)out_size; (void)ws_size;
    const float* src_emb = (const float*)d_in[0];  // [B][D][N]
    const float* tgt_emb = (const float*)d_in[1];  // [B][D][M]
    const float* src     = (const float*)d_in[2];  // [B][3][N]
    const float* tgt     = (const float*)d_in[3];  // [B][3][M]
    // d_in[4] = temperature, unused by the reference math

    const int B = in_sizes[4];
    const int N = in_sizes[2] / (3 * B);
    const int D = in_sizes[0] / (B * N);
    const int M = in_sizes[1] / (B * D);

    char* ws = (char*)d_ws;
    size_t off = 0;
    auto alloc = [&](size_t bytes) {
        size_t o = off;
        off += (bytes + 255) & ~(size_t)255;
        return o;
    };
    float*     Smat = (float*)    (ws + alloc((size_t)B * N * M * 4));
    _Float16*  A16  = (_Float16*) (ws + alloc((size_t)B * N * D * 2));
    _Float16*  B16  = (_Float16*) (ws + alloc((size_t)B * M * D * 2));
    float*     cs   = (float*)    (ws + alloc((size_t)B * N * 4));
    float*     rs   = (float*)    (ws + alloc((size_t)B * M * 4));
    float*     wtg  = (float*)    (ws + alloc((size_t)B * 3 * N * 4));
    float*     wmx  = (float*)    (ws + alloc((size_t)B * N * 4));

    dim3 tb(32, 8, 1);
    k_transpose_f16<<<dim3(N / 32, D / 32, B), tb, 0, stream>>>(src_emb, A16, D, N);
    k_transpose_f16<<<dim3(M / 32, D / 32, B), tb, 0, stream>>>(tgt_emb, B16, D, M);
    k_zero_f32<<<(B * N + 255) / 256, 256, 0, stream>>>(cs, B * N);
    k_zero_f32<<<(B * M + 255) / 256, 256, 0, stream>>>(rs, B * M);

    const float scale = 1.0f / sqrtf((float)D);
    k_gemm_wmma<<<dim3(M / 64, N / 256, B), 256, 0, stream>>>(A16, B16, Smat, N, M, D, scale);

    for (int it = 0; it < 5; ++it) {
        k_rownorm<<<dim3(N, B), 256, 0, stream>>>(Smat, cs, N, M);
        k_colnorm<<<dim3(M / 256, B), 256, 0, stream>>>(Smat, rs, N, M);
    }

    float* outR = (float*)d_out;
    float* outT = outR + (size_t)B * 9;
    float* outP = outT + (size_t)B * 3;
    k_finalize_rows<<<dim3(N, B), 256, 0, stream>>>(Smat, tgt, outP, wtg, wmx, N, M);
    k_svd_head<<<B, 256, 0, stream>>>(src, wtg, wmx, outR, outT, N);
}
